// Predict_89446988907116
// MI455X (gfx1250) — matrix operand order repaired and checked by reading, not compiled, not run on hardware
//
#include <hip/hip_runtime.h>

#define NB        16
#define NBOX      25200
#define NCLS      80
#define KPRE      256
#define MAXB      100
#define TPB       256
#define SCORE_THRF 0.25f
#define IOU_THRF   0.1f

typedef __attribute__((ext_vector_type(16))) _Float16 v16h;
typedef __attribute__((ext_vector_type(8)))  float    v8f;

__constant__ float c_anchors[18] = {10,13, 16,30, 33,23, 30,61, 62,45, 59,119,
                                    116,90, 156,198, 373,326};

__device__ __forceinline__ float sigmoidf_(float x) { return 1.0f / (1.0f + __expf(-x)); }

struct BoxRef { const float* p; int g, a, x, y, abase; };

__device__ __forceinline__ BoxRef resolve_box(const float* p5, const float* p4,
                                              const float* p3, int b, int j) {
  const float* fm; int g, abase, local;
  if (j < 1200)      { fm = p5; g = 20; abase = 6; local = j; }
  else if (j < 6000) { fm = p4; g = 40; abase = 3; local = j - 1200; }
  else               { fm = p3; g = 80; abase = 0; local = j - 6000; }
  int cell = local / 3, a = local - cell * 3;
  int y = cell / g,     x = cell - y * g;
  BoxRef r;
  r.p = fm + (size_t)(((b * g + y) * g + x) * 255 + a * 85);
  r.g = g; r.a = a; r.x = x; r.y = y; r.abase = abase;
  return r;
}

// ---------------------------------------------------------------------------
// Kernel 1: decode all boxes to x1y1x2y2 (pure streaming; warms L2 for topk)
// ---------------------------------------------------------------------------
__global__ void yolo_decode_k(const float* __restrict__ p5, const float* __restrict__ p4,
                              const float* __restrict__ p3, float* __restrict__ boxes) {
  int t = blockIdx.x * TPB + threadIdx.x;          // < NB*NBOX (exact grid)
  int b = t / NBOX, j = t - b * NBOX;
  BoxRef r = resolve_box(p5, p4, p3, b, j);
  float ratio = 640.0f / (float)r.g;
  float cx = (sigmoidf_(r.p[0]) + (float)r.x) * ratio;
  float cy = (sigmoidf_(r.p[1]) + (float)r.y) * ratio;
  float aw = c_anchors[2 * (r.abase + r.a) + 0];
  float ah = c_anchors[2 * (r.abase + r.a) + 1];
  float w = __expf(r.p[2]) * aw;
  float h = __expf(r.p[3]) * ah;
  // warm the conf/class region of this box into L2 for the per-class scans
  __builtin_prefetch(r.p + 4, 0, 3);
  float4 bx = { cx - 0.5f * w, cy - 0.5f * h, cx + 0.5f * w, cy + 0.5f * h };
  ((float4*)boxes)[t] = bx;
}

// ---------------------------------------------------------------------------
// Kernel 2: per (image,class) top-256 via 2-level radix select + bitonic sort.
// Scores recomputed from raw logits (entire activation set resides in 192MB L2)
// ---------------------------------------------------------------------------
__global__ void yolo_topk_k(const float* __restrict__ p5, const float* __restrict__ p4,
                            const float* __restrict__ p3,
                            float* __restrict__ topk_s, int* __restrict__ topk_i) {
  const int blk = blockIdx.x;                 // NB*NCLS blocks
  const int b = blk / NCLS, c = blk - b * NCLS;
  const int tid = threadIdx.x;

  __shared__ unsigned int hist[256];
  __shared__ float cs[KPRE];
  __shared__ int   ci[KPRE];
  __shared__ unsigned int s_cnt, s_thr, s_b0, s_above;
  __shared__ int s_skip;

  // ---- pass 1: coarse histogram on bits[23:16] (scores live in (0.25,1)) ----
  hist[tid] = 0u;
  __syncthreads();
  for (int j = tid; j < NBOX; j += TPB) {
    BoxRef r = resolve_box(p5, p4, p3, b, j);
    float s = sigmoidf_(r.p[4]) * sigmoidf_(r.p[5 + c]);
    if (s > SCORE_THRF) {
      unsigned bits = __float_as_uint(s);
      int bin = (int)(bits >> 16) - 0x3E80; bin = min(max(bin, 0), 255);
      atomicAdd(&hist[bin], 1u);
    }
  }
  __syncthreads();
  if (tid == 0) {
    unsigned acc = 0; int b0 = -1; unsigned above = 0;
    for (int bin = 255; bin >= 0; --bin) {
      unsigned h = hist[bin];
      if (acc + h >= (unsigned)KPRE) { b0 = bin; above = acc; break; }
      acc += h;
    }
    if (b0 < 0) { s_skip = 1; s_thr = 0u; }
    else        { s_skip = 0; s_b0 = (unsigned)b0; s_above = above; }
  }
  __syncthreads();

  // ---- pass 2: refine within boundary bin on bits[15:8] ----
  if (!s_skip) {
    hist[tid] = 0u;
    __syncthreads();
    for (int j = tid; j < NBOX; j += TPB) {
      BoxRef r = resolve_box(p5, p4, p3, b, j);
      float s = sigmoidf_(r.p[4]) * sigmoidf_(r.p[5 + c]);
      if (s > SCORE_THRF) {
        unsigned bits = __float_as_uint(s);
        int bin = (int)(bits >> 16) - 0x3E80; bin = min(max(bin, 0), 255);
        if ((unsigned)bin == s_b0) atomicAdd(&hist[(bits >> 8) & 0xFF], 1u);
      }
    }
    __syncthreads();
    if (tid == 0) {
      unsigned acc = s_above;
      unsigned thr = (0x3E80u + s_b0) << 16;
      for (int m = 255; m >= 0; --m) {
        acc += hist[m];
        if (acc >= (unsigned)KPRE) { thr = ((0x3E80u + s_b0) << 16) | ((unsigned)m << 8); break; }
      }
      s_thr = thr;
    }
    __syncthreads();
  }

  // ---- compact candidates >= threshold (ties beyond 24 bits capped) ----
  if (tid == 0) s_cnt = 0u;
  __syncthreads();
  for (int j = tid; j < NBOX; j += TPB) {
    BoxRef r = resolve_box(p5, p4, p3, b, j);
    float s = sigmoidf_(r.p[4]) * sigmoidf_(r.p[5 + c]);
    if (s > SCORE_THRF && __float_as_uint(s) >= s_thr) {
      unsigned pos = atomicAdd(&s_cnt, 1u);
      if (pos < (unsigned)KPRE) { cs[pos] = s; ci[pos] = j; }
    }
  }
  __syncthreads();
  if (tid >= (int)min(s_cnt, (unsigned)KPRE)) { cs[tid] = -1.0f; ci[tid] = 0; }
  __syncthreads();

  // ---- bitonic sort 256, descending by score ----
  for (int k = 2; k <= KPRE; k <<= 1) {
    for (int j2 = k >> 1; j2 > 0; j2 >>= 1) {
      int ixj = tid ^ j2;
      if (ixj > tid) {
        bool dirDesc = ((tid & k) == 0);
        float a = cs[tid], v = cs[ixj];
        if ((a < v) == dirDesc) {
          cs[tid] = v; cs[ixj] = a;
          int ti = ci[tid]; ci[tid] = ci[ixj]; ci[ixj] = ti;
        }
      }
      __syncthreads();
    }
  }
  topk_s[blk * KPRE + tid] = cs[tid];
  topk_i[blk * KPRE + tid] = ci[tid];
}

// ---------------------------------------------------------------------------
// Kernel 3: greedy NMS per (image,class); cumsum cap done on the matrix unit
// with two exact v_wmma_f32_16x16x32_f16 prefix-sum matmuls.
// ---------------------------------------------------------------------------
__global__ void yolo_nms_k(const float* __restrict__ boxes,
                           const float* __restrict__ topk_s, const int* __restrict__ topk_i,
                           float* __restrict__ nms_s) {
  const int blk = blockIdx.x;
  const int b = blk / NCLS;
  const int tid = threadIdx.x;

  __shared__ float bx1[KPRE], by1[KPRE], bx2[KPRE], by2[KPRE];
  __shared__ int   supp[KPRE];
  __shared__ float kf[KPRE];        // keep flags as f32
  __shared__ float wpre[KPRE];      // within-row inclusive prefix (16x16)
  __shared__ float offs[16];        // per-row offsets

  float s  = topk_s[blk * KPRE + tid];
  int   id = topk_i[blk * KPRE + tid];
  float4 bb = ((const float4*)boxes)[b * NBOX + id];
  bx1[tid] = bb.x; by1[tid] = bb.y; bx2[tid] = bb.z; by2[tid] = bb.w;
  supp[tid] = (s <= SCORE_THRF) ? 1 : 0;
  __syncthreads();

  float myArea = fmaxf(bb.z - bb.x, 0.0f) * fmaxf(bb.w - bb.y, 0.0f);

  for (int i = 0; i < KPRE; ++i) {
    int alive = (supp[i] == 0);
    if (alive && tid > i) {
      float ix1 = bx1[i], iy1 = by1[i], ix2 = bx2[i], iy2 = by2[i];
      float iarea = fmaxf(ix2 - ix1, 0.0f) * fmaxf(iy2 - iy1, 0.0f);
      float iw = fmaxf(fminf(bb.z, ix2) - fmaxf(bb.x, ix1), 0.0f);
      float ih = fmaxf(fminf(bb.w, iy2) - fmaxf(bb.y, iy1), 0.0f);
      float inter = iw * ih;
      float iou = inter / fmaxf(myArea + iarea - inter, 1e-9f);
      if (iou > IOU_THRF) supp[tid] = 1;
    }
    __syncthreads();
  }

  int keep = (supp[tid] == 0);
  kf[tid] = keep ? 1.0f : 0.0f;
  __syncthreads();

  // --- WMMA #1: W = K(16x16 keep flags, padded to 16x32) x U(upper-tri 32x16) ---
  if (tid < 32) {
    int lane = tid, M = lane & 15, half = lane >> 4, N = lane & 15;
    v16h A = {}; v16h Bm = {};
#pragma unroll
    for (int v = 0; v < 4; ++v)
#pragma unroll
      for (int e = 0; e < 2; ++e) {
        int K = half * 8 + 2 * v + e;                 // A 16-bit layout, K<16
        A[2 * v + e] = (_Float16)kf[M * 16 + K];
      }
#pragma unroll
    for (int v = 0; v < 8; ++v)
#pragma unroll
      for (int e = 0; e < 2; ++e) {
        int K = half * 8 + 2 * v + e + (v >= 4 ? 8 : 0); // B 16-bit layout
        Bm[2 * v + e] = (_Float16)((K < 16 && K <= N) ? 1.0f : 0.0f);
      }
    v8f Cz = {};
    v8f D = __builtin_amdgcn_wmma_f32_16x16x32_f16(false, A, false, Bm,
                                                   (short)0, Cz, false, false);
#pragma unroll
    for (int v = 0; v < 8; ++v)
      wpre[(v + 8 * half) * 16 + N] = D[v];
  }
  __syncthreads();

  // --- WMMA #2: row offsets = broadcast(rowTotals) x strictUpper(32x16) ---
  if (tid < 32) {
    int lane = tid, half = lane >> 4, N = lane & 15;
    v16h A2 = {}; v16h B2 = {};
#pragma unroll
    for (int v = 0; v < 4; ++v)
#pragma unroll
      for (int e = 0; e < 2; ++e) {
        int K = half * 8 + 2 * v + e;                 // K<16
        A2[2 * v + e] = (_Float16)wpre[K * 16 + 15];  // t[K] = row total
      }
#pragma unroll
    for (int v = 0; v < 8; ++v)
#pragma unroll
      for (int e = 0; e < 2; ++e) {
        int K = half * 8 + 2 * v + e + (v >= 4 ? 8 : 0);
        B2[2 * v + e] = (_Float16)((K < 16 && K < N) ? 1.0f : 0.0f);
      }
    v8f Cz = {};
    v8f D2 = __builtin_amdgcn_wmma_f32_16x16x32_f16(false, A2, false, B2,
                                                    (short)0, Cz, false, false);
    if (lane < 16) offs[N] = D2[0];                   // all rows identical
  }
  __syncthreads();

  float cum = wpre[tid] + offs[tid >> 4];             // exact inclusive cumsum
  if (cum > (float)MAXB) keep = 0;
  nms_s[blk * KPRE + tid] = keep ? s : -1.0f;
}

// ---------------------------------------------------------------------------
// Kernel 4: per-image merge of 80x256 kept scores -> top-100 output
// ---------------------------------------------------------------------------
__global__ void yolo_final_k(const float* __restrict__ boxes, const int* __restrict__ topk_i,
                             const float* __restrict__ nms_s, float* __restrict__ out) {
  const int b = blockIdx.x;
  const int tid = threadIdx.x;
  const int N2 = NCLS * KPRE;
  const float* src = nms_s + (size_t)b * N2;

  __shared__ unsigned int hist[256];
  __shared__ float fs[128];
  __shared__ int   fidx[128];
  __shared__ unsigned int s_cnt, s_thr, s_b0, s_above;
  __shared__ int s_skip;

  hist[tid] = 0u;
  __syncthreads();
  for (int f = tid; f < N2; f += TPB) {
    __builtin_prefetch(src + f + 8 * TPB, 0, 0);
    float s = src[f];
    if (s > 0.0f) {
      unsigned bits = __float_as_uint(s);
      int bin = (int)(bits >> 16) - 0x3E80; bin = min(max(bin, 0), 255);
      atomicAdd(&hist[bin], 1u);
    }
  }
  __syncthreads();
  if (tid == 0) {
    unsigned acc = 0; int b0 = -1; unsigned above = 0;
    for (int bin = 255; bin >= 0; --bin) {
      unsigned h = hist[bin];
      if (acc + h >= (unsigned)MAXB) { b0 = bin; above = acc; break; }
      acc += h;
    }
    if (b0 < 0) { s_skip = 1; s_thr = 0u; }
    else        { s_skip = 0; s_b0 = (unsigned)b0; s_above = above; }
  }
  __syncthreads();
  if (!s_skip) {
    hist[tid] = 0u;
    __syncthreads();
    for (int f = tid; f < N2; f += TPB) {
      float s = src[f];
      if (s > 0.0f) {
        unsigned bits = __float_as_uint(s);
        int bin = (int)(bits >> 16) - 0x3E80; bin = min(max(bin, 0), 255);
        if ((unsigned)bin == s_b0) atomicAdd(&hist[(bits >> 8) & 0xFF], 1u);
      }
    }
    __syncthreads();
    if (tid == 0) {
      unsigned acc = s_above;
      unsigned thr = (0x3E80u + s_b0) << 16;
      for (int m = 255; m >= 0; --m) {
        acc += hist[m];
        if (acc >= (unsigned)MAXB) { thr = ((0x3E80u + s_b0) << 16) | ((unsigned)m << 8); break; }
      }
      s_thr = thr;
    }
    __syncthreads();
  }

  if (tid == 0) s_cnt = 0u;
  __syncthreads();
  for (int f = tid; f < N2; f += TPB) {
    float s = src[f];
    if (s > 0.0f && __float_as_uint(s) >= s_thr) {
      unsigned pos = atomicAdd(&s_cnt, 1u);
      if (pos < 128u) { fs[pos] = s; fidx[pos] = f; }
    }
  }
  __syncthreads();
  if (tid < 128 && tid >= (int)min(s_cnt, 128u)) { fs[tid] = -1.0f; fidx[tid] = 0; }
  __syncthreads();

  // bitonic sort 128, descending
  for (int k = 2; k <= 128; k <<= 1) {
    for (int j2 = k >> 1; j2 > 0; j2 >>= 1) {
      if (tid < 128) {
        int ixj = tid ^ j2;
        if (ixj > tid) {
          bool dirDesc = ((tid & k) == 0);
          float a = fs[tid], v = fs[ixj];
          if ((a < v) == dirDesc) {
            fs[tid] = v; fs[ixj] = a;
            int ti = fidx[tid]; fidx[tid] = fidx[ixj]; fidx[ixj] = ti;
          }
        }
      }
      __syncthreads();
    }
  }

  float* out_boxes  = out;                                // NB*MAXB*4
  float* out_scores = out + NB * MAXB * 4;                // NB*MAXB
  int*   out_labels = (int*)(out + NB * MAXB * 4 + NB * MAXB);

  if (tid < MAXB) {
    float sc = fs[tid];
    int valid = sc > 0.0f;
    int flat = fidx[tid];
    int label = flat >> 8, slot = flat & 255;
    int boxIdx = topk_i[((size_t)b * NCLS + label) * KPRE + slot];
    float4 bb = ((const float4*)boxes)[b * NBOX + boxIdx];
    int base = (b * MAXB + tid) * 4;
    out_boxes[base + 0] = valid ? bb.x : -1.0f;
    out_boxes[base + 1] = valid ? bb.y : -1.0f;
    out_boxes[base + 2] = valid ? bb.z : -1.0f;
    out_boxes[base + 3] = valid ? bb.w : -1.0f;
    out_scores[b * MAXB + tid] = valid ? sc : -1.0f;
    out_labels[b * MAXB + tid] = valid ? label : -1;
  }
}

// ---------------------------------------------------------------------------
extern "C" void kernel_launch(void* const* d_in, const int* in_sizes, int n_in,
                              void* d_out, int out_size, void* d_ws, size_t ws_size,
                              hipStream_t stream) {
  const float* p5 = (const float*)d_in[0];
  const float* p4 = (const float*)d_in[1];
  const float* p3 = (const float*)d_in[2];

  float* wsf    = (float*)d_ws;
  float* boxes  = wsf;                                  // NB*NBOX*4
  float* topk_s = boxes + (size_t)NB * NBOX * 4;        // NB*NCLS*KPRE
  int*   topk_i = (int*)(topk_s + (size_t)NB * NCLS * KPRE);
  float* nms_s  = (float*)(topk_i + (size_t)NB * NCLS * KPRE);

  yolo_decode_k<<<(NB * NBOX) / TPB, TPB, 0, stream>>>(p5, p4, p3, boxes);
  yolo_topk_k  <<<NB * NCLS, TPB, 0, stream>>>(p5, p4, p3, topk_s, topk_i);
  yolo_nms_k   <<<NB * NCLS, TPB, 0, stream>>>(boxes, topk_s, topk_i, nms_s);
  yolo_final_k <<<NB, TPB, 0, stream>>>(boxes, topk_i, nms_s, (float*)d_out);
}
